// AttentionClassifier_90993177133692
// MI455X (gfx1250) — matrix operand order repaired and checked by reading, not compiled, not run on hardware
//
#include <hip/hip_runtime.h>
#include <cstdint>
#include <cstddef>

// ---------------------------------------------------------------------------
// MI455X (gfx1250) implementation of the BiLSTM + MHA classifier.
// f16 data / f32 accumulate via v_wmma_f32_16x16x32_f16.
//  - big parallel GEMMs: async global->LDS tile staging (ASYNCcnt) double-
//    buffered against ds_load fragment reads + WMMA
//  - recurrent scan: persistent WG per direction, h double-buffered in LDS,
//    c in LDS, W_hh streamed from L2, branch-free gate transcendentals
//  - attention: per-(b,head) block, WMMA scores + LDS softmax + WMMA P@V
// ---------------------------------------------------------------------------

typedef _Float16 f16;
typedef __attribute__((ext_vector_type(16))) _Float16 v16h;
typedef __attribute__((ext_vector_type(8)))  float    v8f;

#define DEV static __device__ __forceinline__

static constexpr int Bsz = 64;
static constexpr int Tsz = 256;
static constexpr int Hsz = 512;
static constexpr int Esz = 256;

DEV int lane_id() { return (int)(threadIdx.x & 31); }

// A fragment: 16x32 f16 tile at `tile` (row stride lda).
// Layout used here (not the ISA table order): lanes 0-15 / 16-31 both hold M=lane&15; K contiguous: k = 16*(lane>>4) + s.
DEV v16h load_a_frag(const f16* tile, int lda) {
  int l = lane_id();
  const f16* row = tile + (size_t)(l & 15) * lda + ((l >> 4) << 4);
  v16h a;
#pragma unroll
  for (int v = 0; v < 8; ++v) {
    int k = 2 * v;
    a[2 * v]     = row[k];
    a[2 * v + 1] = row[k + 1];
  }
  return a;
}

// B fragment: 32x16 f16 tile where element (k,n) = W[n*ldw + k] (row-major
// weight W[N][K]).  Per lane: n = lane&15, 16 contiguous k starting at
// (lane>=16 ? 16 : 0): one contiguous 32-byte run.
DEV v16h load_b_frag(const f16* wtile, int ldw) {
  int l = lane_id();
  const f16* row = wtile + (size_t)(l & 15) * ldw + ((l >> 4) << 4);
  v16h b;
#pragma unroll
  for (int v = 0; v < 16; ++v) b[v] = row[v];
  return b;
}

DEV v8f wmma32(v16h a, v16h b, v8f c) {
  return __builtin_amdgcn_wmma_f32_16x16x32_f16(false, a, false, b,
                                                (short)0, c, false, false);
}

DEV float sigmoidf_(float x) { return 1.0f / (1.0f + __expf(-x)); }
// branch-free tanh: saturates to +/-1 via v_exp_f32, no EXEC divergence
DEV float tanh_fast(float x) {
  float e = __expf(2.0f * x);
  return 1.0f - 2.0f / (e + 1.0f);
}

// ---- CDNA5 async global->LDS copy (ASYNCcnt-tracked) ----------------------
// lds_off: LDS byte address (low 32 bits of a flat LDS pointer, per ISA 10.2
// flat->LDS mapping).  gaddr: 64-bit global address.  16 bytes per lane.
DEV void async_copy_b128(unsigned lds_off, const void* gaddr) {
  asm volatile("global_load_async_to_lds_b128 %0, %1, off"
               :: "v"(lds_off), "v"(gaddr)
               : "memory");
}
DEV void wait_async0() {
#if __has_builtin(__builtin_amdgcn_s_wait_asynccnt)
  __builtin_amdgcn_s_wait_asynccnt(0);
#else
  asm volatile("s_wait_asynccnt 0x0" ::: "memory");
#endif
}

// ---------------------------------------------------------------------------
// WMMA GEMM with async-LDS double-buffered tiles:
//   C[M,N] = A[M,K] * W[N,K]^T + bias[N]
// block = 256 threads (8 waves); block tile 128(M) x 64(N); K-step 32.
// LDS: Abuf[2][128][32] + Bbuf[2][64][32] f16 = 24 KB.
// ---------------------------------------------------------------------------
static constexpr unsigned GEMM_AB = 128 * 32 * sizeof(f16);  // 8192 B per A buf
static constexpr unsigned GEMM_BB = 64 * 32 * sizeof(f16);   // 4096 B per B buf
static constexpr unsigned GEMM_LDS = 2 * GEMM_AB + 2 * GEMM_BB;  // 24576 B

template <typename OUT>
__global__ __launch_bounds__(256) void k_gemm(
    const f16* __restrict__ A, int lda,
    const f16* __restrict__ W, int ldw,
    const float* __restrict__ bias,
    OUT* __restrict__ C, int ldc, int K)
{
  extern __shared__ char smem[];
  const unsigned lds_base = (unsigned)(uintptr_t)(void*)smem;

  const int tid  = (int)threadIdx.x;
  const int wave = tid >> 5;
  const int wm = wave >> 1, wn = wave & 1;
  const size_t mblk = (size_t)blockIdx.y * 128;
  const size_t nblk = (size_t)blockIdx.x * 64;
  const size_t m0 = mblk + (size_t)wm * 32;
  const size_t n0 = nblk + (size_t)wn * 32;

  // issue async copies of the (128x32 A, 64x32 B) tile at k0 into buffer `buf`
  auto stage = [&](int k0, int buf) {
    const unsigned aoff = lds_base + (unsigned)buf * GEMM_AB;
    const unsigned boff = lds_base + 2 * GEMM_AB + (unsigned)buf * GEMM_BB;
#pragma unroll
    for (int c = tid; c < 512; c += 256) {     // A: 128 rows x 4 chunks of 8 f16
      int row = c >> 2, c8 = (c & 3) * 8;
      async_copy_b128(aoff + (unsigned)(row * 32 + c8) * 2,
                      A + (mblk + row) * (size_t)lda + k0 + c8);
    }
    {                                          // B: 64 rows x 2 chunks of 8 f16
      int row = tid >> 1, c8 = (tid & 1) * 8;
      async_copy_b128(boff + (unsigned)(row * 32 + c8) * 2,
                      W + (nblk + row) * (size_t)ldw + k0 + c8);
    }
  };

  v8f acc[2][2] = {};
  const int nk = K / 32;
  stage(0, 0);
  for (int ks = 0; ks < nk; ++ks) {
    wait_async0();          // this wave's issued tile writes are complete
    __syncthreads();        // => all waves' tile-ks writes visible; no wave
                            //    still reads buffer (ks+1)&1 (consumed ks-1)
    if (ks + 1 < nk) stage((ks + 1) * 32, (ks + 1) & 1);

    const f16* Ac = (const f16*)(smem + (size_t)(ks & 1) * GEMM_AB);
    const f16* Bc = (const f16*)(smem + 2 * GEMM_AB + (size_t)(ks & 1) * GEMM_BB);
    v16h a0 = load_a_frag(Ac + (wm * 32 + 0)  * 32, 32);
    v16h a1 = load_a_frag(Ac + (wm * 32 + 16) * 32, 32);
    v16h b0 = load_b_frag(Bc + (wn * 32 + 0)  * 32, 32);
    v16h b1 = load_b_frag(Bc + (wn * 32 + 16) * 32, 32);
    acc[0][0] = wmma32(a0, b0, acc[0][0]);
    acc[0][1] = wmma32(a0, b1, acc[0][1]);
    acc[1][0] = wmma32(a1, b0, acc[1][0]);
    acc[1][1] = wmma32(a1, b1, acc[1][1]);
  }

  const int l = lane_id();
  const int nc = l & 15, mb = (l >> 4) << 3;
#pragma unroll
  for (int i = 0; i < 2; ++i)
#pragma unroll
    for (int j = 0; j < 2; ++j) {
      size_t mm = m0 + (size_t)i * 16 + mb;
      size_t nn = n0 + (size_t)j * 16 + nc;
      float bv = bias ? bias[nn] : 0.0f;
#pragma unroll
      for (int r = 0; r < 8; ++r)
        C[(mm + r) * (size_t)ldc + nn] = (OUT)(acc[i][j][r] + bv);
    }
}

// ---------------------------------------------------------------------------
// Small helper kernels
// ---------------------------------------------------------------------------
__global__ __launch_bounds__(256) void k_cvt(const float* __restrict__ s,
                                             f16* __restrict__ d, int n) {
  int i = (int)(blockIdx.x * blockDim.x + threadIdx.x);
  if (i < n) d[i] = (f16)s[i];
}

__global__ __launch_bounds__(256) void k_bias2(const float* __restrict__ a,
                                               const float* __restrict__ b,
                                               float* __restrict__ d, int n) {
  int i = (int)(blockIdx.x * blockDim.x + threadIdx.x);
  if (i < n) d[i] = a[i] + b[i];
}

__global__ __launch_bounds__(256) void k_embed(const int* __restrict__ inp,
                                               const float* __restrict__ emb,
                                               f16* __restrict__ x0) {
  size_t i = (size_t)blockIdx.x * blockDim.x + threadIdx.x;  // over B*T*E
  size_t bt = i >> 8;                                        // E == 256
  int e = (int)(i & 255);
  int tok = inp[bt];
  x0[i] = (f16)emb[(size_t)tok * Esz + e];
}

__global__ __launch_bounds__(256) void k_sumdirs(const f16* __restrict__ x2,
                                                 f16* __restrict__ xs) {
  size_t i = (size_t)blockIdx.x * blockDim.x + threadIdx.x;  // over B*T*H
  size_t bt = i >> 9;
  int h = (int)(i & 511);
  xs[i] = (f16)((float)x2[bt * 1024 + h] + (float)x2[bt * 1024 + 512 + h]);
}

// ---------------------------------------------------------------------------
// Bidirectional LSTM recurrent scan.  grid.x = 2 (fwd, bwd), 512 threads
// (16 waves).  LDS: h double-buffer [2][64][512] f16 + c [64][512] f32 =
// 256 KB of the WGP's 320 KB.
// ---------------------------------------------------------------------------
__global__ __launch_bounds__(512) void k_lstm_scan(
    const f16* __restrict__ gpre,   // [2][B*T][2048]  (x@Wih^T + b_ih + b_hh)
    const f16* __restrict__ whh0,   // [2048][512] f16
    const f16* __restrict__ whh1,
    f16* __restrict__ xout)         // [B*T][1024]; dir0 -> [:512], dir1 -> [512:]
{
  extern __shared__ char smem[];
  f16*   hbuf = (f16*)smem;                                   // [2][64][512]
  float* cst  = (float*)(smem + 2 * 64 * 512 * sizeof(f16));  // [64][512]

  const int dir = (int)blockIdx.x;
  const f16* whh = dir ? whh1 : whh0;
  const f16* gp  = gpre + (size_t)dir * ((size_t)Bsz * Tsz * 2048);

  const int tid = (int)threadIdx.x;
  for (int i = tid; i < 2 * 64 * 512; i += 512) hbuf[i] = (f16)0.0f;
  for (int i = tid; i < 64 * 512; i += 512) cst[i] = 0.0f;
  __syncthreads();

  const int wave = tid >> 5;
  const int l = tid & 31;
  const int nc = l & 15, mb = (l >> 4) << 3;

  for (int t = 0; t < Tsz; ++t) {
    const int t_act = dir ? (Tsz - 1 - t) : t;
    const f16* hcur = hbuf + (size_t)(t & 1) * (64 * 512);
    f16*       hnxt = hbuf + (size_t)((t + 1) & 1) * (64 * 512);

#pragma unroll 1
    for (int j = 0; j < 8; ++j) {
      const int q  = wave * 8 + j;
      const int tm = q >> 5;       // batch tile (0..3)
      const int hn = q & 31;       // hidden chunk (0..31)
      v8f acc[4] = {};
      for (int k0 = 0; k0 < 512; k0 += 32) {
        v16h a = load_a_frag(hcur + (size_t)tm * 16 * 512 + k0, 512);
#pragma unroll
        for (int g = 0; g < 4; ++g) {
          v16h b = load_b_frag(whh + (size_t)(g * 512 + hn * 16) * 512 + k0, 512);
          acc[g] = wmma32(a, b, acc[g]);
        }
      }
      const int h_idx = hn * 16 + nc;
#pragma unroll
      for (int r = 0; r < 8; ++r) {
        const int b = tm * 16 + mb + r;
        const f16* g4 = gp + ((size_t)b * Tsz + t_act) * 2048;
        float gi = acc[0][r] + (float)g4[h_idx];
        float gf = acc[1][r] + (float)g4[512 + h_idx];
        float gg = acc[2][r] + (float)g4[1024 + h_idx];
        float go = acc[3][r] + (float)g4[1536 + h_idx];
        float iv = sigmoidf_(gi);
        float fv = sigmoidf_(gf);
        float gv = tanh_fast(gg);
        float ov = sigmoidf_(go);
        float c  = cst[b * 512 + h_idx];
        float cn = fv * c + iv * gv;
        cst[b * 512 + h_idx] = cn;
        float hv = ov * tanh_fast(cn);
        hnxt[b * 512 + h_idx] = (f16)hv;
        xout[((size_t)b * Tsz + t_act) * 1024 + dir * 512 + h_idx] = (f16)hv;
      }
    }
    __syncthreads();   // new-h writes visible before next step's reads
  }
}

// ---------------------------------------------------------------------------
// Attention: one block per (b, head).  256 threads (8 waves).
// LDS: S [256][256] f16 (128KB) + Vt [64][256] f16 (32KB).
// ---------------------------------------------------------------------------
__global__ __launch_bounds__(256) void k_attn(const f16* __restrict__ qkv,
                                              f16* __restrict__ octx)
{
  extern __shared__ char smem[];
  f16* S  = (f16*)smem;                               // [256][256]
  f16* Vt = (f16*)(smem + 256 * 256 * sizeof(f16));   // [64][256]

  const int blk = (int)blockIdx.x;        // b*8 + head
  const int b = blk >> 3, head = blk & 7;
  const f16* base = qkv + (size_t)b * Tsz * 1536;
  const int qoff = head * 64;
  const int koff = 512 + head * 64;
  const int voff = 1024 + head * 64;

  const int tid = (int)threadIdx.x;
  for (int i = tid; i < 64 * 256; i += 256) {
    int d = i >> 8, tp = i & 255;
    Vt[d * 256 + tp] = base[(size_t)tp * 1536 + voff + d];
  }
  __syncthreads();

  const int wave = tid >> 5, l = tid & 31;
  const int nc = l & 15, mb = (l >> 4) << 3;

  // scores S = (q * 1/sqrt(hd)) @ k^T
#pragma unroll 1
  for (int j = 0; j < 32; ++j) {
    int tile = wave * 32 + j;
    int tm = tile >> 4, tn = tile & 15;
    v8f acc = {};
    for (int k0 = 0; k0 < 64; k0 += 32) {
      v16h a  = load_a_frag(base + (size_t)tm * 16 * 1536 + qoff + k0, 1536);
      v16h bf = load_b_frag(base + (size_t)tn * 16 * 1536 + koff + k0, 1536);
      acc = wmma32(a, bf, acc);
    }
#pragma unroll
    for (int r = 0; r < 8; ++r)
      S[(tm * 16 + mb + r) * 256 + tn * 16 + nc] = (f16)(acc[r] * 0.125f);
  }
  __syncthreads();

  // softmax: one row per thread, in place (thread-local row, no hazards)
  {
    const int r = tid;
    float mx = -1e30f;
    for (int jx = 0; jx < 256; ++jx) mx = fmaxf(mx, (float)S[r * 256 + jx]);
    float sum = 0.0f;
    for (int jx = 0; jx < 256; ++jx) {
      float e = __expf((float)S[r * 256 + jx] - mx);
      sum += e;
      S[r * 256 + jx] = (f16)e;
    }
    float inv = 1.0f / sum;
    for (int jx = 0; jx < 256; ++jx)
      S[r * 256 + jx] = (f16)((float)S[r * 256 + jx] * inv);
  }
  __syncthreads();

  // O = P @ V  (M=256, N=64, K=256) ; B fragments from transposed V in LDS
#pragma unroll 1
  for (int j = 0; j < 8; ++j) {
    int tile = wave * 8 + j;
    int tm = tile >> 2, tn = tile & 3;
    v8f acc = {};
    for (int k0 = 0; k0 < 256; k0 += 32) {
      v16h a  = load_a_frag(S + (size_t)tm * 16 * 256 + k0, 256);
      v16h bf = load_b_frag(Vt + (size_t)tn * 16 * 256 + k0, 256);
      acc = wmma32(a, bf, acc);
    }
#pragma unroll
    for (int r = 0; r < 8; ++r)
      octx[((size_t)b * Tsz + tm * 16 + mb + r) * 512 + head * 64 + tn * 16 + nc]
          = (f16)acc[r];
  }
}

// ---------------------------------------------------------------------------
// Final: out[b] = dot(oproj[b, :], out_w) + out_b   (131072 elements / row)
// ---------------------------------------------------------------------------
__global__ __launch_bounds__(256) void k_final(const float* __restrict__ oproj,
                                               const float* __restrict__ ow,
                                               const float* __restrict__ ob,
                                               float* __restrict__ out)
{
  __shared__ float red[256];
  const int b = (int)blockIdx.x, tid = (int)threadIdx.x;
  const float* row = oproj + (size_t)b * (Tsz * Hsz);
  float s = 0.0f;
  for (int i = tid; i < Tsz * Hsz; i += 256) s += row[i] * ow[i];
  red[tid] = s;
  __syncthreads();
  for (int st = 128; st > 0; st >>= 1) {
    if (tid < st) red[tid] += red[tid + st];
    __syncthreads();
  }
  if (tid == 0) out[b] = red[0] + ob[0];
}

// ---------------------------------------------------------------------------
// Host orchestration
// ---------------------------------------------------------------------------
extern "C" void kernel_launch(void* const* d_in, const int* in_sizes, int n_in,
                              void* d_out, int out_size, void* d_ws, size_t ws_size,
                              hipStream_t stream) {
  (void)in_sizes; (void)n_in; (void)out_size; (void)ws_size;
  char* ws = (char*)d_ws;

  const int*   inp = (const int*)d_in[0];
  const float* emb = (const float*)d_in[1];
  const float* wih_f[4] = {(const float*)d_in[2], (const float*)d_in[3],
                           (const float*)d_in[4], (const float*)d_in[5]};
  const float* whh_f[4] = {(const float*)d_in[6], (const float*)d_in[7],
                           (const float*)d_in[8], (const float*)d_in[9]};
  const float* bih_f[4] = {(const float*)d_in[10], (const float*)d_in[11],
                           (const float*)d_in[12], (const float*)d_in[13]};
  const float* bhh_f[4] = {(const float*)d_in[14], (const float*)d_in[15],
                           (const float*)d_in[16], (const float*)d_in[17]};
  const float* aiw = (const float*)d_in[18];
  const float* aib = (const float*)d_in[19];
  const float* aow = (const float*)d_in[20];
  const float* aob = (const float*)d_in[21];
  const float* ow  = (const float*)d_in[22];
  const float* ob  = (const float*)d_in[23];

  // workspace layout (deterministic; aligned to 256B)
  size_t off = 0;
  auto alloc = [&](size_t bytes) -> size_t {
    size_t o = off;
    off = (off + bytes + 255) & ~(size_t)255;
    return o;
  };
  const size_t M = (size_t)Bsz * Tsz;   // 16384
  size_t oWIH[4], oWHH[4], oBC[4];
  const size_t wih_elems[4] = {2048 * 256, 2048 * 256, 2048 * 1024, 2048 * 1024};
  for (int i = 0; i < 4; ++i) oWIH[i] = alloc(wih_elems[i] * sizeof(f16));
  for (int i = 0; i < 4; ++i) oWHH[i] = alloc((size_t)2048 * 512 * sizeof(f16));
  for (int i = 0; i < 4; ++i) oBC[i]  = alloc((size_t)2048 * sizeof(float));
  size_t oAIW  = alloc((size_t)1536 * 512 * sizeof(f16));
  size_t oAOW  = alloc((size_t)512 * 512 * sizeof(f16));
  size_t oX0   = alloc(M * Esz * sizeof(f16));
  size_t oX1   = alloc(M * 1024 * sizeof(f16));   // layer0 out; later OPROJ (f32)
  size_t oX2   = alloc(M * 1024 * sizeof(f16));   // layer1 out; later OCTX
  size_t oXS   = alloc(M * 512 * sizeof(f16));
  size_t oGPRE = alloc((size_t)2 * M * 2048 * sizeof(f16));  // also QKV

  f16*   WIH[4]; f16* WHH[4]; float* BC[4];
  for (int i = 0; i < 4; ++i) { WIH[i] = (f16*)(ws + oWIH[i]); WHH[i] = (f16*)(ws + oWHH[i]); BC[i] = (float*)(ws + oBC[i]); }
  f16*   AIW   = (f16*)(ws + oAIW);
  f16*   AOW   = (f16*)(ws + oAOW);
  f16*   X0    = (f16*)(ws + oX0);
  f16*   X1    = (f16*)(ws + oX1);
  f16*   X2    = (f16*)(ws + oX2);
  f16*   XS    = (f16*)(ws + oXS);
  f16*   GPRE  = (f16*)(ws + oGPRE);
  f16*   QKV   = GPRE;                 // alias (layer gpre dead by then)
  f16*   OCTX  = X2;                   // alias (x2 dead after sumdirs)
  float* OPROJ = (float*)(ws + oX1);   // alias (x1 dead after layer1 gemms)
  const size_t gstride = M * 2048;     // per-direction gpre stride (elements)

  auto cvt = [&](const float* src, f16* dst, size_t n) {
    k_cvt<<<dim3((unsigned)((n + 255) / 256)), dim3(256), 0, stream>>>(src, dst, (int)n);
  };

  // 1) weights -> f16, combined biases
  for (int i = 0; i < 4; ++i) cvt(wih_f[i], WIH[i], wih_elems[i]);
  for (int i = 0; i < 4; ++i) cvt(whh_f[i], WHH[i], (size_t)2048 * 512);
  for (int i = 0; i < 4; ++i)
    k_bias2<<<dim3(8), dim3(256), 0, stream>>>(bih_f[i], bhh_f[i], BC[i], 2048);
  cvt(aiw, AIW, (size_t)1536 * 512);
  cvt(aow, AOW, (size_t)512 * 512);

  // 2) embedding gather
  k_embed<<<dim3((unsigned)(M * Esz / 256)), dim3(256), 0, stream>>>(inp, emb, X0);

  const size_t scan_lds = (size_t)2 * 64 * 512 * sizeof(f16) + (size_t)64 * 512 * sizeof(float);
  const size_t attn_lds = (size_t)256 * 256 * sizeof(f16) + (size_t)64 * 256 * sizeof(f16);

  // 3) layer 0: input GEMMs (K=256, N=2048) then scan
  {
    dim3 g(2048 / 64, (unsigned)(M / 128));
    k_gemm<f16><<<g, dim3(256), GEMM_LDS, stream>>>(X0, 256, WIH[0], 256, BC[0], GPRE, 2048, 256);
    k_gemm<f16><<<g, dim3(256), GEMM_LDS, stream>>>(X0, 256, WIH[1], 256, BC[1], GPRE + gstride, 2048, 256);
    k_lstm_scan<<<dim3(2), dim3(512), scan_lds, stream>>>(GPRE, WHH[0], WHH[1], X1);
  }
  // 4) layer 1: input GEMMs (K=1024) then scan
  {
    dim3 g(2048 / 64, (unsigned)(M / 128));
    k_gemm<f16><<<g, dim3(256), GEMM_LDS, stream>>>(X1, 1024, WIH[2], 1024, BC[2], GPRE, 2048, 1024);
    k_gemm<f16><<<g, dim3(256), GEMM_LDS, stream>>>(X1, 1024, WIH[3], 1024, BC[3], GPRE + gstride, 2048, 1024);
    k_lstm_scan<<<dim3(2), dim3(512), scan_lds, stream>>>(GPRE, WHH[2], WHH[3], X2);
  }

  // 5) sum directions -> XS [B*T, 512]
  k_sumdirs<<<dim3((unsigned)(M * 512 / 256)), dim3(256), 0, stream>>>(X2, XS);

  // 6) QKV GEMM (N=1536, K=512) into GPRE region
  {
    dim3 g(1536 / 64, (unsigned)(M / 128));
    k_gemm<f16><<<g, dim3(256), GEMM_LDS, stream>>>(XS, 512, AIW, 512, aib, QKV, 1536, 512);
  }

  // 7) attention per (b, head)
  k_attn<<<dim3(Bsz * 8), dim3(256), attn_lds, stream>>>(QKV, OCTX);

  // 8) out projection (N=512, K=512), f32 output
  {
    dim3 g(512 / 64, (unsigned)(M / 128));
    k_gemm<float><<<g, dim3(256), GEMM_LDS, stream>>>(OCTX, 512, AOW, 512, aob, OPROJ, 512, 512);
  }

  // 9) final reduction -> d_out [B,1] (f32)
  k_final<<<dim3(Bsz), dim3(256), 0, stream>>>(OPROJ, ow, ob, (float*)d_out);
}
